// MSDeformAttnPixelDecoder_6347961664025
// MI455X (gfx1250) — matrix operand order repaired and checked by reading, not compiled, not run on hardware
//
#include <hip/hip_runtime.h>

typedef __attribute__((ext_vector_type(16))) _Float16 v16h;
typedef __attribute__((ext_vector_type(8)))  _Float16 v8h;
typedef __attribute__((ext_vector_type(4)))  _Float16 v4h;
typedef __attribute__((ext_vector_type(8)))  float    v8f;
typedef __attribute__((ext_vector_type(4)))  float    v4f;

#define QTOT  5376
#define BATCH 4
#define MROWS 21504      // BATCH * QTOT
#define DMODEL 256

template<class X, class Y> struct is_same_t { static constexpr bool value = false; };
template<class X> struct is_same_t<X, X>    { static constexpr bool value = true;  };

// ---------------------------------------------------------------------------
// prep: (B,256,H,W) -> x[b, start+hw, c]  and posq = pos + level_embed
// ---------------------------------------------------------------------------
__global__ __launch_bounds__(256) void prep_kernel(
    const float* __restrict__ src, const float* __restrict__ pos,
    const float* __restrict__ lemb, float* __restrict__ x, float* __restrict__ pq,
    int HW, int start)
{
  __shared__ float ts[32][33];
  __shared__ float tp[32][33];
  int hw0 = blockIdx.x * 32, c0 = blockIdx.y * 32, b = blockIdx.z;
  int tx = threadIdx.x, ty = threadIdx.y;
  #pragma unroll
  for (int i = 0; i < 4; ++i) {
    int c = c0 + ty + i * 8;
    size_t o = ((size_t)b * DMODEL + c) * HW + hw0 + tx;
    ts[ty + i * 8][tx] = src[o];
    tp[ty + i * 8][tx] = pos[o];
  }
  __syncthreads();
  #pragma unroll
  for (int i = 0; i < 4; ++i) {
    int hw = hw0 + ty + i * 8;
    int c  = c0 + tx;
    size_t o = ((size_t)b * QTOT + start + hw) * DMODEL + c;
    x[o]  = ts[tx][ty + i * 8];
    pq[o] = tp[tx][ty + i * 8] + lemb[c];
  }
}

// ---------------------------------------------------------------------------
// fp32 -> f16 weight conversion (one time)
// ---------------------------------------------------------------------------
__global__ __launch_bounds__(256) void f32_to_f16_kernel(
    const float* __restrict__ in, _Float16* __restrict__ out, int n)
{
  int idx = (blockIdx.x * 256 + threadIdx.x) * 4;
  if (idx < n) {
    v4f v = *(const v4f*)(in + idx);
    v4h h;
    #pragma unroll
    for (int k = 0; k < 4; ++k) h[k] = (_Float16)v[k];
    *(v4h*)(out + idx) = h;
  }
}

// ---------------------------------------------------------------------------
// WMMA fragment loaders (contiguous k order, same in A and B, wave32)
// A 16x32 f16: lane<16 -> kbase 0, lane>=16 -> kbase 16; elems 0-15 = kbase..+15
//              (two 16B LDS loads)
// B 32x16 f16: lane n = lane&15; elems = 16 contiguous K at (lane>>4)*16
// ---------------------------------------------------------------------------
__device__ __forceinline__ v16h load_frag_a(const _Float16* As, int lane, int ms, int ldk)
{
  int row = ms + (lane & 15);
  int kb  = (lane >> 4) * 16;
  const _Float16* p = As + row * ldk + kb;
  v8h lo = *(const v8h*)p;
  v8h hi = *(const v8h*)(p + 8);
  return __builtin_shufflevector(lo, hi, 0,1,2,3,4,5,6,7,8,9,10,11,12,13,14,15);
}

__device__ __forceinline__ v16h load_frag_b(const _Float16* Ws, int lane, int ns, int ldk)
{
  int row = ns + (lane & 15);
  int kb  = (lane >> 4) * 16;
  const _Float16* p = Ws + row * ldk + kb;
  v8h lo = *(const v8h*)p;
  v8h hi = *(const v8h*)(p + 8);
  return __builtin_shufflevector(lo, hi, 0,1,2,3,4,5,6,7,8,9,10,11,12,13,14,15);
}

// ---------------------------------------------------------------------------
// GEMM: C[M,N] = act( (A1 [+A2]) @ W^T + bias ),  W is f16 row-major N x K
// block = 256 threads (8 waves), block tile 128x64, wave tile 32x32, K-step 32
// M assumed multiple of 128, K multiple of 32; N guarded.
// ---------------------------------------------------------------------------
template<typename InT, typename OutT, bool RELU, bool ADD2>
__global__ __launch_bounds__(256) void gemm_kernel(
    const InT* __restrict__ A, const float* __restrict__ A2,
    const _Float16* __restrict__ W, const float* __restrict__ bias,
    OutT* __restrict__ C, int N, int K)
{
  constexpr int BM = 128, BN = 64, BK = 32, LDK = 40;
  __shared__ __attribute__((aligned(16))) _Float16 As[BM * LDK];
  __shared__ __attribute__((aligned(16))) _Float16 Ws[BN * LDK];

  int tid  = threadIdx.x;
  int lane = tid & 31;
  int wave = tid >> 5;
  int m0 = blockIdx.x * BM;
  int n0 = blockIdx.y * BN;
  int wm = (wave >> 1) * 32;   // 0,32,64,96
  int wn = (wave & 1) * 32;    // 0,32

  v8f acc[2][2] = {};

  for (int k0 = 0; k0 < K; k0 += BK) {
    // ---- stage A tile (fp32 or f16 -> f16 in LDS) -----------------------
    {
      int r = tid >> 3;           // 0..31
      int c = (tid & 7) * 4;      // 0..28
      #pragma unroll
      for (int p = 0; p < 4; ++p) {
        int row = r + p * 32;
        size_t go = (size_t)(m0 + row) * K + k0 + c;
        float f[4];
        if constexpr (is_same_t<InT, float>::value) {
          v4f v = *(const v4f*)(A + go);
          #pragma unroll
          for (int k = 0; k < 4; ++k) f[k] = v[k];
          if constexpr (ADD2) {
            v4f v2 = *(const v4f*)(A2 + go);
            #pragma unroll
            for (int k = 0; k < 4; ++k) f[k] += v2[k];
          }
        } else {
          v4h v = *(const v4h*)(A + go);
          #pragma unroll
          for (int k = 0; k < 4; ++k) f[k] = (float)v[k];
        }
        v4h h;
        #pragma unroll
        for (int k = 0; k < 4; ++k) h[k] = (_Float16)f[k];
        *(v4h*)&As[row * LDK + c] = h;
      }
    }
    // ---- stage W tile ----------------------------------------------------
    {
      int r = tid >> 2;           // 0..63
      int c = (tid & 3) * 8;      // 0,8,16,24
      int n = n0 + r;
      v8h v = {};
      if (n < N) v = *(const v8h*)(W + (size_t)n * K + k0 + c);
      *(v8h*)&Ws[r * LDK + c] = v;
    }
    __syncthreads();

    v16h af0 = load_frag_a(As, lane, wm,      LDK);
    v16h af1 = load_frag_a(As, lane, wm + 16, LDK);
    v16h bf0 = load_frag_b(Ws, lane, wn,      LDK);
    v16h bf1 = load_frag_b(Ws, lane, wn + 16, LDK);

    acc[0][0] = __builtin_amdgcn_wmma_f32_16x16x32_f16(false, af0, false, bf0, (short)0, acc[0][0], false, false);
    acc[0][1] = __builtin_amdgcn_wmma_f32_16x16x32_f16(false, af0, false, bf1, (short)0, acc[0][1], false, false);
    acc[1][0] = __builtin_amdgcn_wmma_f32_16x16x32_f16(false, af1, false, bf0, (short)0, acc[1][0], false, false);
    acc[1][1] = __builtin_amdgcn_wmma_f32_16x16x32_f16(false, af1, false, bf1, (short)0, acc[1][1], false, false);

    __syncthreads();
  }

  // ---- epilogue: bias (+relu), store. D layout: vgpr r -> m = 8*(l>>4)+r ---
  #pragma unroll
  for (int j = 0; j < 2; ++j) {
    int n = n0 + wn + j * 16 + (lane & 15);
    bool ok = n < N;
    float bv = ok ? bias[n] : 0.0f;
    #pragma unroll
    for (int i = 0; i < 2; ++i) {
      int mb = m0 + wm + i * 16 + ((lane >> 4) << 3);
      #pragma unroll
      for (int r = 0; r < 8; ++r) {
        float v = acc[i][j][r] + bv;
        if (RELU) v = v > 0.0f ? v : 0.0f;
        if (ok) C[(size_t)(mb + r) * N + n] = (OutT)v;
      }
    }
  }
}

// ---------------------------------------------------------------------------
// MSDA sampling: one wave32 per (b,q,head); lane = channel within head (HD=32)
// ---------------------------------------------------------------------------
__device__ __forceinline__ float corner_val(const float* vb, int xi, int yi, int Wl, int Hl)
{
  if (xi < 0 || xi >= Wl || yi < 0 || yi >= Hl) return 0.0f;
  return vb[(size_t)(yi * Wl + xi) * DMODEL];
}

__global__ __launch_bounds__(256) void sample_kernel(
    const float* __restrict__ value, const float* __restrict__ off,
    const float* __restrict__ aw, float* __restrict__ attn)
{
  int gw   = (blockIdx.x * 256 + threadIdx.x) >> 5;   // 0 .. MROWS*8-1
  int lane = threadIdx.x & 31;
  int h  = gw & 7;
  int bq = gw >> 3;                  // b*QTOT + q
  int b  = bq / QTOT;
  int q  = bq - b * QTOT;

  int Hq, Wq, iq;
  if (q < 4096)      { Hq = 64; Wq = 64; iq = q; }
  else if (q < 5120) { Hq = 32; Wq = 32; iq = q - 4096; }
  else               { Hq = 16; Wq = 16; iq = q - 5120; }
  float rx = ((float)(iq % Wq) + 0.5f) / (float)Wq;
  float ry = ((float)(iq / Wq) + 0.5f) / (float)Hq;

  // softmax over 12 attention weights of this head
  const float* ap = aw + (size_t)bq * 96 + h * 12;
  float a[12], mx = -1e30f;
  #pragma unroll
  for (int j = 0; j < 12; ++j) { a[j] = ap[j]; mx = fmaxf(mx, a[j]); }
  float s = 0.0f;
  #pragma unroll
  for (int j = 0; j < 12; ++j) { a[j] = __expf(a[j] - mx); s += a[j]; }
  float inv = 1.0f / s;

  const float* op = off + (size_t)bq * 192 + h * 24;

  const int starts[3] = {0, 4096, 5120};
  const int dims[3]   = {64, 32, 16};

  float acc = 0.0f;
  #pragma unroll
  for (int l = 0; l < 3; ++l) {
    int Wl = dims[l], Hl = dims[l];
    const float* vb = value + ((size_t)b * QTOT + starts[l]) * DMODEL + h * 32 + lane;
    float invW = 1.0f / (float)Wl, invH = 1.0f / (float)Hl;
    #pragma unroll
    for (int p = 0; p < 4; ++p) {
      float ox = op[(l * 4 + p) * 2 + 0];
      float oy = op[(l * 4 + p) * 2 + 1];
      float xf = (rx + ox * invW) * (float)Wl - 0.5f;
      float yf = (ry + oy * invH) * (float)Hl - 0.5f;
      float x0f = floorf(xf), y0f = floorf(yf);
      float fx = xf - x0f,    fy = yf - y0f;
      int x0 = (int)x0f, y0 = (int)y0f;
      float wgt = a[l * 4 + p] * inv;
      float v = corner_val(vb, x0,     y0,     Wl, Hl) * (1.0f - fx) * (1.0f - fy)
              + corner_val(vb, x0 + 1, y0,     Wl, Hl) * fx          * (1.0f - fy)
              + corner_val(vb, x0,     y0 + 1, Wl, Hl) * (1.0f - fx) * fy
              + corner_val(vb, x0 + 1, y0 + 1, Wl, Hl) * fx          * fy;
      acc += wgt * v;
    }
  }
  attn[(size_t)bq * DMODEL + h * 32 + lane] = acc;
}

// ---------------------------------------------------------------------------
// fused residual add + LayerNorm: x = LN(x + r) * g + b ; one wave32 per row
// ---------------------------------------------------------------------------
__global__ __launch_bounds__(256) void add_ln_kernel(
    float* __restrict__ x, const float* __restrict__ r,
    const float* __restrict__ g, const float* __restrict__ b)
{
  int row  = (blockIdx.x * 256 + threadIdx.x) >> 5;
  int lane = threadIdx.x & 31;
  float* xr = x + (size_t)row * DMODEL;
  const float* rr = r + (size_t)row * DMODEL;
  int c0 = lane * 4, c1 = 128 + lane * 4;
  v4f a0 = *(const v4f*)(xr + c0);
  v4f r0 = *(const v4f*)(rr + c0);
  v4f a1 = *(const v4f*)(xr + c1);
  v4f r1 = *(const v4f*)(rr + c1);
  float v[8];
  #pragma unroll
  for (int k = 0; k < 4; ++k) { v[k] = a0[k] + r0[k]; v[4 + k] = a1[k] + r1[k]; }
  float s = 0.0f, s2 = 0.0f;
  #pragma unroll
  for (int k = 0; k < 8; ++k) { s += v[k]; s2 += v[k] * v[k]; }
  #pragma unroll
  for (int o = 16; o > 0; o >>= 1) {
    s  += __shfl_xor(s,  o, 32);
    s2 += __shfl_xor(s2, o, 32);
  }
  float mean = s * (1.0f / 256.0f);
  float var  = s2 * (1.0f / 256.0f) - mean * mean;
  float rstd = rsqrtf(var + 1e-5f);
  #pragma unroll
  for (int k = 0; k < 4; ++k) xr[c0 + k] = (v[k]     - mean) * rstd * g[c0 + k] + b[c0 + k];
  #pragma unroll
  for (int k = 0; k < 4; ++k) xr[c1 + k] = (v[4 + k] - mean) * rstd * g[c1 + k] + b[c1 + k];
}

// ---------------------------------------------------------------------------
// host orchestration
// ---------------------------------------------------------------------------
extern "C" void kernel_launch(void* const* d_in, const int* in_sizes, int n_in,
                              void* d_out, int out_size, void* d_ws, size_t ws_size,
                              hipStream_t stream)
{
  (void)in_sizes; (void)n_in; (void)out_size; (void)ws_size;
  const float* src[3] = { (const float*)d_in[0], (const float*)d_in[1], (const float*)d_in[2] };
  const float* pos[3] = { (const float*)d_in[3], (const float*)d_in[4], (const float*)d_in[5] };
  const float* lemb   = (const float*)d_in[6];
  const float* off_w  = (const float*)d_in[7];
  const float* off_b  = (const float*)d_in[8];
  const float* aw_w   = (const float*)d_in[9];
  const float* aw_b   = (const float*)d_in[10];
  const float* val_w  = (const float*)d_in[11];
  const float* val_b  = (const float*)d_in[12];
  const float* out_w  = (const float*)d_in[13];
  const float* out_b  = (const float*)d_in[14];
  const float* ln1_g  = (const float*)d_in[15];
  const float* ln1_b  = (const float*)d_in[16];
  const float* ffn1_w = (const float*)d_in[17];
  const float* ffn1_b = (const float*)d_in[18];
  const float* ffn2_w = (const float*)d_in[19];
  const float* ffn2_b = (const float*)d_in[20];
  const float* ln2_g  = (const float*)d_in[21];
  const float* ln2_b  = (const float*)d_in[22];

  char* ws = (char*)d_ws;
  size_t cur = 0;
  auto alloc = [&](size_t bytes) -> char* {
    char* p = ws + cur;
    cur += (bytes + 255) & ~(size_t)255;
    return p;
  };
  float*    X    = (float*)alloc((size_t)MROWS * 256 * 4);
  float*    PQ   = (float*)alloc((size_t)MROWS * 256 * 4);
  float*    VAL  = (float*)alloc((size_t)MROWS * 256 * 4);
  float*    OFF  = (float*)alloc((size_t)MROWS * 192 * 4);
  float*    AW   = (float*)alloc((size_t)MROWS * 96 * 4);
  float*    ATT  = (float*)alloc((size_t)MROWS * 256 * 4);
  float*    TMP  = (float*)alloc((size_t)MROWS * 256 * 4);
  _Float16* Hbuf = (_Float16*)alloc((size_t)MROWS * 1024 * 2);
  _Float16* Woff = (_Float16*)alloc((size_t)6 * 192 * 256 * 2);
  _Float16* Waw  = (_Float16*)alloc((size_t)6 * 96 * 256 * 2);
  _Float16* Wval = (_Float16*)alloc((size_t)6 * 256 * 256 * 2);
  _Float16* Wout = (_Float16*)alloc((size_t)6 * 256 * 256 * 2);
  _Float16* Wf1  = (_Float16*)alloc((size_t)6 * 1024 * 256 * 2);
  _Float16* Wf2  = (_Float16*)alloc((size_t)6 * 256 * 1024 * 2);

  // ---- prep: flatten + transpose + level embed ----
  {
    dim3 tb(32, 8);
    const int HWs[3]    = {4096, 1024, 256};
    const int startv[3] = {0, 4096, 5120};
    for (int l = 0; l < 3; ++l) {
      dim3 gr(HWs[l] / 32, 256 / 32, BATCH);
      prep_kernel<<<gr, tb, 0, stream>>>(src[l], pos[l], lemb + l * 256, X, PQ, HWs[l], startv[l]);
    }
  }

  // ---- convert all weights to f16 once ----
  auto cvt = [&](const float* in, _Float16* out, int n) {
    f32_to_f16_kernel<<<(n / 4 + 255) / 256, 256, 0, stream>>>(in, out, n);
  };
  cvt(off_w,  Woff, 6 * 192 * 256);
  cvt(aw_w,   Waw,  6 * 96 * 256);
  cvt(val_w,  Wval, 6 * 256 * 256);
  cvt(out_w,  Wout, 6 * 256 * 256);
  cvt(ffn1_w, Wf1,  6 * 1024 * 256);
  cvt(ffn2_w, Wf2,  6 * 256 * 1024);

  const int GM = MROWS / 128;   // 168

  for (int i = 0; i < 6; ++i) {
    // value = x @ val_w^T + val_b
    gemm_kernel<float, float, false, false><<<dim3(GM, 4), 256, 0, stream>>>(
        X, nullptr, Wval + (size_t)i * 256 * 256, val_b + (size_t)i * 256, VAL, 256, 256);
    // off = (x+pos) @ off_w^T + off_b
    gemm_kernel<float, float, false, true><<<dim3(GM, 3), 256, 0, stream>>>(
        X, PQ, Woff + (size_t)i * 192 * 256, off_b + (size_t)i * 192, OFF, 192, 256);
    // aw = (x+pos) @ aw_w^T + aw_b
    gemm_kernel<float, float, false, true><<<dim3(GM, 2), 256, 0, stream>>>(
        X, PQ, Waw + (size_t)i * 96 * 256, aw_b + (size_t)i * 96, AW, 96, 256);
    // deformable sampling (softmax + bilinear gather)
    sample_kernel<<<MROWS, 256, 0, stream>>>(VAL, OFF, AW, ATT);
    // out-proj
    gemm_kernel<float, float, false, false><<<dim3(GM, 4), 256, 0, stream>>>(
        ATT, nullptr, Wout + (size_t)i * 256 * 256, out_b + (size_t)i * 256, TMP, 256, 256);
    // x = LN1(x + attn)
    add_ln_kernel<<<MROWS / 8, 256, 0, stream>>>(X, TMP, ln1_g + (size_t)i * 256, ln1_b + (size_t)i * 256);
    // h = relu(x @ ffn1^T + b)  (f16 output)
    gemm_kernel<float, _Float16, true, false><<<dim3(GM, 16), 256, 0, stream>>>(
        X, nullptr, Wf1 + (size_t)i * 1024 * 256, ffn1_b + (size_t)i * 1024, Hbuf, 1024, 256);
    // y = h @ ffn2^T + b
    gemm_kernel<_Float16, float, false, false><<<dim3(GM, 4), 256, 0, stream>>>(
        Hbuf, nullptr, Wf2 + (size_t)i * 256 * 1024, ffn2_b + (size_t)i * 256, TMP, 256, 1024);
    // x = LN2(x + y)
    add_ln_kernel<<<MROWS / 8, 256, 0, stream>>>(X, TMP, ln2_g + (size_t)i * 256, ln2_b + (size_t)i * 256);
  }

  hipMemcpyAsync(d_out, X, (size_t)MROWS * 256 * 4, hipMemcpyDeviceToDevice, stream);
}
